// SelfAttention_31920196943874
// MI455X (gfx1250) — compile-verified, matrix operand order checked by reading, not run on hardware
//
#include <hip/hip_runtime.h>
#include <hip/hip_bf16.h>

#define B_ 4
#define S_ 2048
#define E_ 1024
#define H_ 16
#define D_ 64

typedef __attribute__((ext_vector_type(16))) __bf16 v16bf;
typedef __attribute__((ext_vector_type(8)))  __bf16 bf16x8;
typedef __attribute__((ext_vector_type(8)))  float  v8f;

// ---------------------------------------------------------------------------
// D = A(16x32 bf16) x B(32x16 bf16) + C(16x16 f32), wave32 WMMA
// ---------------------------------------------------------------------------
static __device__ inline v8f wmma_bf16(v16bf a, v16bf b, v8f c) {
  return __builtin_amdgcn_wmma_f32_16x16x32_bf16(
      /*neg_a=*/false, a, /*neg_b=*/false, b,
      /*c_mod=*/(short)0, c, /*reuse_a=*/false, /*reuse_b=*/false);
}

// A/B fragment loader (16x32, 16-bit elements), ISA 7.12.2 layout:
// lane l holds row (l&15); VGPRs 0..3 = K (l>>4)*8 + 0..7, VGPRs 4..7 =
// K 16 + (l>>4)*8 + 0..7.  `p` points at row base + (l>>4)*8; we read
// p[0..7] and p[16..23] (two 16B chunks -> b128 loads, global or ds).
static __device__ inline v16bf frag_from_bf16(const __bf16* p) {
  bf16x8 lo = *(const bf16x8*)(p);
  bf16x8 hi = *(const bf16x8*)(p + 16);
  v16bf f;
#pragma unroll
  for (int i = 0; i < 8; ++i) { f[i] = lo[i]; f[i + 8] = hi[i]; }
  return f;
}

// Row-wise reductions across the 16-lane half-wave (C-layout rows live in a
// fixed half; xor masks <=8 never cross halves).
static __device__ inline float redmax16(float v) {
  v = fmaxf(v, __shfl_xor(v, 1, 32));
  v = fmaxf(v, __shfl_xor(v, 2, 32));
  v = fmaxf(v, __shfl_xor(v, 4, 32));
  v = fmaxf(v, __shfl_xor(v, 8, 32));
  return v;
}
static __device__ inline float redsum16(float v) {
  v += __shfl_xor(v, 1, 32);
  v += __shfl_xor(v, 2, 32);
  v += __shfl_xor(v, 4, 32);
  v += __shfl_xor(v, 8, 32);
  return v;
}

// CDNA5 async copy: memory -> LDS, tracked by ASYNCcnt.
static __device__ inline void async_b128_to_lds(const __bf16* gptr, const __bf16* lptr) {
  unsigned loff = (unsigned)(uintptr_t)lptr;               // low 32 bits = LDS offset
  unsigned long long ga = (unsigned long long)(uintptr_t)gptr;
  asm volatile("global_load_async_to_lds_b128 %0, %1, off"
               :: "v"(loff), "v"(ga) : "memory");
}

// ---------------------------------------------------------------------------
// Kernel 0: one-shot f32 -> bf16 conversion (8 elements / thread).
// ---------------------------------------------------------------------------
__global__ __launch_bounds__(256) void cvt_bf16_kernel(
    const float* __restrict__ src, __bf16* __restrict__ dst, int n8)
{
  const int i = blockIdx.x * blockDim.x + threadIdx.x;
  if (i >= n8) return;
  const float4* s = (const float4*)src + (size_t)i * 2;
  const float4 a = s[0], b = s[1];
  bf16x8 o;
  o[0] = (__bf16)a.x; o[1] = (__bf16)a.y; o[2] = (__bf16)a.z; o[3] = (__bf16)a.w;
  o[4] = (__bf16)b.x; o[5] = (__bf16)b.y; o[6] = (__bf16)b.z; o[7] = (__bf16)b.w;
  ((bf16x8*)dst)[i] = o;
}

// ---------------------------------------------------------------------------
// Kernel 1: fused Q/K/V projection, bf16 in / bf16 out.  Each wave computes a
// 32x64 C macro-tile (2 A-frags x 4 B-frags -> 8 WMMAs per k-step).  Block =
// 4 waves = 64 rows x 128 cols.  grid = (M/64, E/128, 3).
// Q,K stored (B,H,S,D); V stored transposed (B,H,D,S).
// ---------------------------------------------------------------------------
__global__ __launch_bounds__(128) void qkv_proj_kernel(
    const __bf16* __restrict__ xb,  const __bf16* __restrict__ Wqb,
    const __bf16* __restrict__ Wkb, const __bf16* __restrict__ Wvb,
    __bf16* __restrict__ Qb, __bf16* __restrict__ Kb, __bf16* __restrict__ Vtb)
{
  const int lane = threadIdx.x & 31;
  const int wv   = threadIdx.x >> 5;
  const int row0 = blockIdx.x * 64  + (wv >> 1) * 32;
  const int col0 = blockIdx.y * 128 + (wv & 1) * 64;
  const int z    = blockIdx.z;
  const __bf16* W = (z == 0) ? Wqb : (z == 1) ? Wkb : Wvb;

  const int lr  = lane & 15;
  const int kh8 = (lane >> 4) << 3;
  const __bf16* pa0 = xb + (size_t)(row0 + lr) * E_ + kh8;
  const __bf16* pa1 = pa0 + (size_t)16 * E_;
  const __bf16* pb  = W  + (size_t)(col0 + lr) * E_ + kh8;

  v8f c[2][4] = {};
#pragma unroll 2
  for (int kk = 0; kk < E_; kk += 32) {
    __builtin_prefetch(pa0 + kk + 256, 0, 1);
    __builtin_prefetch(pb + kk + 256, 0, 1);
    const v16bf a0 = frag_from_bf16(pa0 + kk);
    const v16bf a1 = frag_from_bf16(pa1 + kk);
#pragma unroll
    for (int nt = 0; nt < 4; ++nt) {
      const v16bf b = frag_from_bf16(pb + (size_t)nt * 16 * E_ + kk);
      c[0][nt] = wmma_bf16(a0, b, c[0][nt]);
      c[1][nt] = wmma_bf16(a1, b, c[1][nt]);
    }
  }

  const int n  = lane & 15;
  const int mh = (lane >> 4) << 3;
#pragma unroll
  for (int mt = 0; mt < 2; ++mt) {
#pragma unroll
    for (int nt = 0; nt < 4; ++nt) {
      const int j  = col0 + nt * 16 + n;
      const int hh = j >> 6, dd = j & (D_ - 1);
#pragma unroll
      for (int v = 0; v < 8; ++v) {
        const int r  = row0 + mt * 16 + v + mh;   // global row in (B*S)
        const int bb = r >> 11;                   // S_ == 2048
        const int s  = r & (S_ - 1);
        const __bf16 val = (__bf16)c[mt][nt][v];
        if (z == 2) {
          Vtb[(((size_t)bb * H_ + hh) * D_ + dd) * S_ + s] = val;
        } else {
          __bf16* dst = (z == 0) ? Qb : Kb;
          dst[(((size_t)bb * H_ + hh) * S_ + s) * D_ + dd] = val;
        }
      }
    }
  }
}

// ---------------------------------------------------------------------------
// Kernel 2: flash-attention per (b,h).  Block = 4 waves = 64 query rows.
// K (32x64) and V^T (64x32) tiles are staged through DOUBLE-BUFFERED LDS via
// global_load_async_to_lds_b128: block kb+1 is issued before computing block
// kb, and s_wait_asynccnt 0x4 drains only the older 4 per-thread transfers
// (async loads complete in order), overlapping DMA with WMMA + softmax.
// grid = (S/64, H, B), block = 128.
// ---------------------------------------------------------------------------
__global__ __launch_bounds__(128) void attn_kernel(
    const __bf16* __restrict__ Qb, const __bf16* __restrict__ Kb,
    const __bf16* __restrict__ Vtb, const unsigned char* __restrict__ pm,
    __bf16* __restrict__ ao)
{
  __shared__ __bf16 ldsK[2][32][64];   // K rows (key, d)      2 x 4 KB
  __shared__ __bf16 ldsV[2][64][32];   // V^T rows (d, key)    2 x 4 KB
  __shared__ __bf16 ldsP[4][16][32];   // per-wave P tile          4 KB

  const int lane = threadIdx.x & 31;
  const int wv   = threadIdx.x >> 5;
  const int tid  = threadIdx.x;
  const int qblk = blockIdx.x;
  const int h    = blockIdx.y;
  const int bb   = blockIdx.z;
  const int row0 = qblk * 64 + wv * 16;

  const __bf16* Qh = Qb  + ((size_t)bb * H_ + h) * (size_t)S_ * D_;
  const __bf16* Kh = Kb  + ((size_t)bb * H_ + h) * (size_t)S_ * D_;
  const __bf16* Vh = Vtb + ((size_t)bb * H_ + h) * (size_t)D_ * S_;
  const unsigned char* pmb = pm + (size_t)bb * S_;

  const int lr  = lane & 15;
  const int kh8 = (lane >> 4) << 3;
  const int n   = lane & 15;
  const int mh  = (lane >> 4) << 3;

  // per-thread chunk coordinates for cooperative K/V staging
  const int kr0 = tid >> 3,         kc0 = (tid & 7) * 8;          // K chunks 0..127
  const int kr1 = (tid + 128) >> 3, kc1 = ((tid + 128) & 7) * 8;  // K chunks 128..255
  const int vr0 = tid >> 2,         vc0 = (tid & 3) * 8;          // V chunks 0..127
  const int vr1 = (tid + 128) >> 2, vc1 = ((tid + 128) & 3) * 8;  // V chunks 128..255

  // Q fragments for the whole head dim (K = 64 -> 2 steps), loaded once.
  const v16bf aq0 = frag_from_bf16(Qh + (size_t)(row0 + lr) * D_ + kh8);
  const v16bf aq1 = frag_from_bf16(Qh + (size_t)(row0 + lr) * D_ + 32 + kh8);

  v8f o0 = {}, o1 = {}, o2 = {}, o3 = {};
  float m8[8], l8[8];
#pragma unroll
  for (int v = 0; v < 8; ++v) { m8[v] = -3.0e38f; l8[v] = 0.0f; }

  const int nkb = (qblk + 1) * 2;      // key blocks of 32 up to causal edge

  // prologue: stage block 0 into buffer 0 (4 async transfers / thread)
  {
    const int key0 = 0;
    async_b128_to_lds(Kh + (size_t)(key0 + kr0) * D_ + kc0, &ldsK[0][kr0][kc0]);
    async_b128_to_lds(Kh + (size_t)(key0 + kr1) * D_ + kc1, &ldsK[0][kr1][kc1]);
    async_b128_to_lds(Vh + (size_t)vr0 * S_ + key0 + vc0, &ldsV[0][vr0][vc0]);
    async_b128_to_lds(Vh + (size_t)vr1 * S_ + key0 + vc1, &ldsV[0][vr1][vc1]);
  }

  for (int kb = 0; kb < nkb; ++kb) {
    const int key0 = kb * 32;
    const int buf  = kb & 1;

    // ---- issue async staging for the NEXT key block (other buffer) ----
    {
      const int knx = ((kb + 1 < nkb) ? kb + 1 : kb) * 32;
      const int nbf = (kb + 1) & 1;
      async_b128_to_lds(Kh + (size_t)(knx + kr0) * D_ + kc0, &ldsK[nbf][kr0][kc0]);
      async_b128_to_lds(Kh + (size_t)(knx + kr1) * D_ + kc1, &ldsK[nbf][kr1][kc1]);
      async_b128_to_lds(Vh + (size_t)vr0 * S_ + knx + vc0, &ldsV[nbf][vr0][vc0]);
      async_b128_to_lds(Vh + (size_t)vr1 * S_ + knx + vc1, &ldsV[nbf][vr1][vc1]);
    }
    // drain only the OLDER 4 transfers (current block); next block stays in flight
    asm volatile("s_wait_asynccnt 0x4" ::: "memory");
    __syncthreads();

    // ---- scores S(16 x 32 keys) = Q K^T  (B-frags from LDS) ----
    v8f s0 = {}, s1 = {};
    s0 = wmma_bf16(aq0, frag_from_bf16(&ldsK[buf][lr][kh8]), s0);
    s0 = wmma_bf16(aq1, frag_from_bf16(&ldsK[buf][lr][32 + kh8]), s0);
    s1 = wmma_bf16(aq0, frag_from_bf16(&ldsK[buf][16 + lr][kh8]), s1);
    s1 = wmma_bf16(aq1, frag_from_bf16(&ldsK[buf][16 + lr][32 + kh8]), s1);

    // ---- scale + causal + key-padding mask ----
    const int kj0 = key0 + n, kj1 = key0 + 16 + n;
    const bool pad0 = pmb[kj0] != 0;
    const bool pad1 = pmb[kj1] != 0;
#pragma unroll
    for (int v = 0; v < 8; ++v) {
      const int qi = row0 + v + mh;
      float a = s0[v] * 0.125f;        // 1/sqrt(64)
      float b = s1[v] * 0.125f;
      if (pad0 || kj0 > qi) a = -1.0e30f;
      if (pad1 || kj1 > qi) b = -1.0e30f;
      s0[v] = a; s1[v] = b;
    }

    // ---- online softmax update ----
#pragma unroll
    for (int v = 0; v < 8; ++v) {
      const float mx    = redmax16(fmaxf(s0[v], s1[v]));
      const float mnew  = fmaxf(m8[v], mx);
      const float alpha = __expf(m8[v] - mnew);
      m8[v] = mnew;
      const float p0 = __expf(s0[v] - mnew);
      const float p1 = __expf(s1[v] - mnew);
      l8[v] = l8[v] * alpha + redsum16(p0 + p1);
      o0[v] *= alpha; o1[v] *= alpha; o2[v] *= alpha; o3[v] *= alpha;
      ldsP[wv][v + mh][n]      = (__bf16)p0;   // C-layout -> row-major tile
      ldsP[wv][v + mh][16 + n] = (__bf16)p1;
    }

    // ---- O += P V  (A from per-wave LDS tile, B from shared V^T tile) ----
    const v16bf ap = frag_from_bf16(&ldsP[wv][lr][kh8]);
    o0 = wmma_bf16(ap, frag_from_bf16(&ldsV[buf][0  + lr][kh8]), o0);
    o1 = wmma_bf16(ap, frag_from_bf16(&ldsV[buf][16 + lr][kh8]), o1);
    o2 = wmma_bf16(ap, frag_from_bf16(&ldsV[buf][32 + lr][kh8]), o2);
    o3 = wmma_bf16(ap, frag_from_bf16(&ldsV[buf][48 + lr][kh8]), o3);

    __syncthreads();   // all waves done reading buf before it is re-staged
  }

  // ---- normalize + store attention output in (B,S,E) bf16 ----
#pragma unroll
  for (int v = 0; v < 8; ++v) {
    const float inv = l8[v] > 0.0f ? 1.0f / l8[v] : 0.0f;
    const size_t base = ((size_t)bb * S_ + (row0 + v + mh)) * E_ + (size_t)h * D_;
    ao[base + 0  + n] = (__bf16)(o0[v] * inv);
    ao[base + 16 + n] = (__bf16)(o1[v] * inv);
    ao[base + 32 + n] = (__bf16)(o2[v] * inv);
    ao[base + 48 + n] = (__bf16)(o3[v] * inv);
  }
  // dangling async transfers for the speculative last stage are drained by
  // s_endpgm's implicit wait-idle.
}

// ---------------------------------------------------------------------------
// Kernel 3: out = attn @ Wo^T + bo, zero padded query rows, f32 output.
// Same 32x64-per-wave blocking.  grid = (M/64, E/128), block = 128.
// ---------------------------------------------------------------------------
__global__ __launch_bounds__(128) void out_proj_kernel(
    const __bf16* __restrict__ ao, const __bf16* __restrict__ Wob,
    const float* __restrict__ bo, const unsigned char* __restrict__ pm,
    float* __restrict__ out)
{
  const int lane = threadIdx.x & 31;
  const int wv   = threadIdx.x >> 5;
  const int row0 = blockIdx.x * 64  + (wv >> 1) * 32;
  const int col0 = blockIdx.y * 128 + (wv & 1) * 64;
  const int lr   = lane & 15;
  const int kh8  = (lane >> 4) << 3;

  const __bf16* pa0 = ao  + (size_t)(row0 + lr) * E_ + kh8;
  const __bf16* pa1 = pa0 + (size_t)16 * E_;
  const __bf16* pb  = Wob + (size_t)(col0 + lr) * E_ + kh8;

  v8f c[2][4] = {};
#pragma unroll 2
  for (int kk = 0; kk < E_; kk += 32) {
    __builtin_prefetch(pa0 + kk + 256, 0, 1);
    __builtin_prefetch(pb + kk + 256, 0, 1);
    const v16bf a0 = frag_from_bf16(pa0 + kk);
    const v16bf a1 = frag_from_bf16(pa1 + kk);
#pragma unroll
    for (int nt = 0; nt < 4; ++nt) {
      const v16bf b = frag_from_bf16(pb + (size_t)nt * 16 * E_ + kk);
      c[0][nt] = wmma_bf16(a0, b, c[0][nt]);
      c[1][nt] = wmma_bf16(a1, b, c[1][nt]);
    }
  }

  const int n  = lane & 15;
  const int mh = (lane >> 4) << 3;
#pragma unroll
  for (int mt = 0; mt < 2; ++mt) {
#pragma unroll
    for (int nt = 0; nt < 4; ++nt) {
      const float bias = bo[col0 + nt * 16 + n];
#pragma unroll
      for (int v = 0; v < 8; ++v) {
        const int r = row0 + mt * 16 + v + mh;   // flat (B*S) row
        const float val = pm[r] ? 0.0f : (c[mt][nt][v] + bias);
        out[(size_t)r * E_ + col0 + nt * 16 + n] = val;
      }
    }
  }
}

// ---------------------------------------------------------------------------
extern "C" void kernel_launch(void* const* d_in, const int* in_sizes, int n_in,
                              void* d_out, int out_size, void* d_ws, size_t ws_size,
                              hipStream_t stream) {
  const float* x          = (const float*)d_in[0];
  const unsigned char* pm = (const unsigned char*)d_in[1];  // bool mask (B,S)
  const float* Wq         = (const float*)d_in[2];
  const float* Wk         = (const float*)d_in[3];
  const float* Wv         = (const float*)d_in[4];
  const float* Wo         = (const float*)d_in[5];
  const float* bo         = (const float*)d_in[6];
  float* out              = (float*)d_out;

  const size_t xe = (size_t)B_ * S_ * E_;           // 8,388,608 elements
  const size_t we = (size_t)E_ * E_;                // 1,048,576 elements
  char* ws = (char*)d_ws;
  size_t off = 0;
  __bf16* xb  = (__bf16*)(ws + off); off += 2 * xe; // 16 MB
  __bf16* Wqb = (__bf16*)(ws + off); off += 2 * we; //  2 MB
  __bf16* Wkb = (__bf16*)(ws + off); off += 2 * we;
  __bf16* Wvb = (__bf16*)(ws + off); off += 2 * we;
  __bf16* Wob = (__bf16*)(ws + off); off += 2 * we;
  __bf16* Qb  = (__bf16*)(ws + off); off += 2 * xe; // (B,H,S,D)
  __bf16* Kb  = (__bf16*)(ws + off); off += 2 * xe; // (B,H,S,D)
  __bf16* Vtb = (__bf16*)(ws + off); off += 2 * xe; // (B,H,D,S)
  __bf16* ao  = (__bf16*)(ws + off); off += 2 * xe; // (B,S,E)

  // one-shot f32 -> bf16 conversions
  cvt_bf16_kernel<<<dim3((unsigned)(xe / 8 / 256)), dim3(256), 0, stream>>>(x, xb, (int)(xe / 8));
  cvt_bf16_kernel<<<dim3((unsigned)(we / 8 / 256)), dim3(256), 0, stream>>>(Wq, Wqb, (int)(we / 8));
  cvt_bf16_kernel<<<dim3((unsigned)(we / 8 / 256)), dim3(256), 0, stream>>>(Wk, Wkb, (int)(we / 8));
  cvt_bf16_kernel<<<dim3((unsigned)(we / 8 / 256)), dim3(256), 0, stream>>>(Wv, Wvb, (int)(we / 8));
  cvt_bf16_kernel<<<dim3((unsigned)(we / 8 / 256)), dim3(256), 0, stream>>>(Wo, Wob, (int)(we / 8));

  const dim3 blk(128);
  qkv_proj_kernel<<<dim3((B_ * S_) / 64, E_ / 128, 3), blk, 0, stream>>>(
      xb, Wqb, Wkb, Wvb, Qb, Kb, Vtb);
  attn_kernel<<<dim3(S_ / 64, H_, B_), blk, 0, stream>>>(Qb, Kb, Vtb, pm, ao);
  out_proj_kernel<<<dim3((B_ * S_) / 64, E_ / 128), blk, 0, stream>>>(
      ao, Wob, bo, pm, out);
}
